// Transformer_GNN_57234734186916
// MI455X (gfx1250) — matrix operand order repaired and checked by reading, compiled, not run on hardware
//
#include <hip/hip_runtime.h>
#include <hip/hip_bf16.h>
#include <math.h>

// ---------------------------------------------------------------------------
// Types for CDNA5 WMMA
// ---------------------------------------------------------------------------
typedef __attribute__((ext_vector_type(16))) __bf16 v16bf;
typedef __attribute__((ext_vector_type(8)))  __bf16 v8bf;
typedef __attribute__((ext_vector_type(8)))  float  v8f;

#define NN    12800
#define EE    102400
#define GG    128
#define DH    256
#define HH    4
#define CC    64
#define DEE   768
#define DOUT  10

static __device__ __forceinline__ v16bf tgnn_concat8(v8bf lo, v8bf hi) {
    return __builtin_shufflevector(lo, hi, 0, 1, 2, 3, 4, 5, 6, 7,
                                           8, 9, 10, 11, 12, 13, 14, 15);
}

// ---------------------------------------------------------------------------
// WMMA GEMM: C[M x Nout] = A[M x K] @ W[K x Nout] (+bias) (optional ReLU)
//   A  : bf16, row-major [M x K]
//   Wt : bf16, TRANSPOSED weights, row-major [Nout x K]
// Block = 256 threads = 8 wave32s. Block tile: 32 (M) x 128 (N), K-step 64.
// Each wave owns one 16-col sub-tile and two 16-row sub-tiles:
// 4 x v_wmma_f32_16x16x32_bf16 per K-step per wave.
// Requires: M%32==0, K%64==0, Nout%128==0 (true for all call sites).
// ---------------------------------------------------------------------------
#define SAPAD 72   // LDS row stride (bf16 elems); 72*2B=144B spreads banks
__global__ __launch_bounds__(256)
void tgnn_gemm_wmma(const __bf16* __restrict__ A, const __bf16* __restrict__ Wt,
                    const float* __restrict__ bias, float* __restrict__ C,
                    int M, int K, int Nout, int relu) {
    __shared__ __align__(16) __bf16 sA[32 * SAPAD];    // A tile  32 x 64
    __shared__ __align__(16) __bf16 sB[128 * SAPAD];   // B^T tile 128 x 64

    const int tid  = threadIdx.x;
    const int lane = tid & 31;
    const int wv   = tid >> 5;        // wave 0..7 -> N sub-tile
    const int half = lane >> 4;       // lane group 0-15 / 16-31
    const int rl   = lane & 15;

    const int m0 = blockIdx.x * 32;
    const int n0 = blockIdx.y * 128;

    // A staging: one 8-elem (16B) chunk per thread: row = tid/8, col8 = (tid%8)*8
    const int ar = tid >> 3;
    const int ac = (tid & 7) * 8;

    v8f acc0 = {}, acc1 = {};

    for (int k0 = 0; k0 < K; k0 += 64) {
        // ---- stage A tile (32 x 64): b128 load -> b128 LDS store ----
        *(v8bf*)&sA[ar * SAPAD + ac] =
            *(const v8bf*)&A[(size_t)(m0 + ar) * K + k0 + ac];
        // ---- stage B^T tile (128 cols x 64 k): 4 chunks per thread ----
        #pragma unroll
        for (int j = 0; j < 4; ++j) {
            int cid = tid + j * 256;
            int c  = cid >> 3;
            int k8 = (cid & 7) * 8;
            *(v8bf*)&sB[c * SAPAD + k8] =
                *(const v8bf*)&Wt[(size_t)(n0 + c) * K + k0 + k8];
        }
        if (k0 + 64 < K) {
            __builtin_prefetch(&A[(size_t)(m0 + ar) * K + k0 + 64 + ac], 0, 3);
        }
        __syncthreads();

        // ---- two 16x16x32 WMMA steps per K-step, 2 M sub-tiles each ----
        #pragma unroll
        for (int kk = 0; kk < 64; kk += 32) {
            // B fragment: lane-half selects K 0-7/16-23 vs 8-15/24-31,
            // column N = wv*16 + rl  (contiguous in B^T LDS)
            const __bf16* pb = &sB[(wv * 16 + rl) * SAPAD + kk + half * 8];
            v16bf fb = tgnn_concat8(*(const v8bf*)pb, *(const v8bf*)(pb + 16));
            // A fragments: row M = rl; lane-half selects K 0-7/16-23 vs 8-15/24-31
            const __bf16* pa0 = &sA[rl * SAPAD + kk + half * 8];
            v16bf fa0 = tgnn_concat8(*(const v8bf*)pa0, *(const v8bf*)(pa0 + 16));
            const __bf16* pa1 = &sA[(16 + rl) * SAPAD + kk + half * 8];
            v16bf fa1 = tgnn_concat8(*(const v8bf*)pa1, *(const v8bf*)(pa1 + 16));

            acc0 = __builtin_amdgcn_wmma_f32_16x16x32_bf16(
                false, fa0, false, fb, (short)0, acc0, false, false);
            acc1 = __builtin_amdgcn_wmma_f32_16x16x32_bf16(
                false, fa1, false, fb, (short)0, acc1, false, false);
        }
        __syncthreads();
    }

    // ---- epilogue: lane<16 -> rows 0-7, lane>=16 -> rows 8-15 of each tile ----
    const int col = n0 + wv * 16 + rl;
    const float bv = bias ? bias[col] : 0.0f;
    #pragma unroll
    for (int i = 0; i < 8; ++i) {
        float y0 = acc0[i] + bv;
        float y1 = acc1[i] + bv;
        if (relu) { y0 = fmaxf(y0, 0.0f); y1 = fmaxf(y1, 0.0f); }
        C[(size_t)(m0 + half * 8 + i) * Nout + col]      = y0;
        C[(size_t)(m0 + 16 + half * 8 + i) * Nout + col] = y1;
    }
}

// ---------------------------------------------------------------------------
// Conversion kernels (f32 -> bf16; weights also transposed to N-major)
// ---------------------------------------------------------------------------
__global__ void tgnn_conv_bf16(const float* __restrict__ in,
                               __bf16* __restrict__ out, int n) {
    int t = blockIdx.x * blockDim.x + threadIdx.x;
    if (t < n) out[t] = (__bf16)in[t];
}

__global__ void tgnn_convtr(const float* __restrict__ W, __bf16* __restrict__ Wt,
                            int K, int Nout) {
    int t = blockIdx.x * blockDim.x + threadIdx.x;
    if (t >= K * Nout) return;
    int k = t / Nout, n = t - k * Nout;
    Wt[(size_t)n * K + k] = (__bf16)W[t];
}

// ---------------------------------------------------------------------------
// Elementwise / attention / reduction kernels
// ---------------------------------------------------------------------------
__global__ void tgnn_fill(float* __restrict__ p, float v, int n) {
    int t = blockIdx.x * blockDim.x + threadIdx.x;
    if (t < n) p[t] = v;
}

// alpha[e*H+h] = (q[dst] . (k[src] + eproj[e])) / sqrt(C)
__global__ void tgnn_alpha(const float* __restrict__ q, const float* __restrict__ k,
                           const float* __restrict__ ep,
                           const int* __restrict__ src, const int* __restrict__ dst,
                           float* __restrict__ alpha, int E) {
    int t = blockIdx.x * blockDim.x + threadIdx.x;
    if (t >= E * HH) return;
    int e = t >> 2, h = t & 3;
    int s = src[e], d = dst[e];
    const float4* qp = (const float4*)(q + (size_t)d * DH + h * CC);
    const float4* kp = (const float4*)(k + (size_t)s * DH + h * CC);
    const float4* pp = (const float4*)(ep + (size_t)e * DH + h * CC);
    float acc = 0.0f;
    #pragma unroll
    for (int c = 0; c < CC / 4; ++c) {
        float4 a = qp[c], b = kp[c], g = pp[c];
        acc += a.x * (b.x + g.x) + a.y * (b.y + g.y)
             + a.z * (b.z + g.z) + a.w * (b.w + g.w);
    }
    alpha[t] = acc * 0.125f; // 1/sqrt(64)
}

__device__ __forceinline__ void tgnn_atomic_max_f32(float* addr, float val) {
    int* ai = (int*)addr;
    int old = *ai;
    while (__int_as_float(old) < val) {
        int assumed = old;
        old = atomicCAS(ai, assumed, __float_as_int(val));
        if (old == assumed) break;
    }
}

__global__ void tgnn_segmax(const float* __restrict__ alpha, const int* __restrict__ dst,
                            float* __restrict__ amax, int E) {
    int t = blockIdx.x * blockDim.x + threadIdx.x;
    if (t >= E * HH) return;
    int e = t >> 2, h = t & 3;
    tgnn_atomic_max_f32(&amax[dst[e] * HH + h], alpha[t]);
}

__global__ void tgnn_expsum(float* __restrict__ alpha, const int* __restrict__ dst,
                            const float* __restrict__ amax, float* __restrict__ ssum, int E) {
    int t = blockIdx.x * blockDim.x + threadIdx.x;
    if (t >= E * HH) return;
    int e = t >> 2, h = t & 3;
    float a = __expf(alpha[t] - amax[dst[e] * HH + h]);
    alpha[t] = a;
    atomicAdd(&ssum[dst[e] * HH + h], a);
}

// out[dst] += (alpha/sum) * (v[src] + eproj[e]) — one block per edge
__global__ __launch_bounds__(256)
void tgnn_scatter(const float* __restrict__ v, const float* __restrict__ ep,
                  const float* __restrict__ alpha, const float* __restrict__ ssum,
                  const int* __restrict__ src, const int* __restrict__ dst,
                  float* __restrict__ out) {
    int e = blockIdx.x;
    int d = threadIdx.x;
    int h = d >> 6;
    int s = src[e], n = dst[e];
    float coef = alpha[e * HH + h] / (ssum[n * HH + h] + 1e-16f);
    float val  = coef * (v[(size_t)s * DH + d] + ep[(size_t)e * DH + d]);
    atomicAdd(&out[(size_t)n * DH + d], val);
}

// per-column mean/var over N rows; one block per column
__global__ __launch_bounds__(256)
void tgnn_bnstats(const float* __restrict__ x, float* __restrict__ mu,
                  float* __restrict__ var, int n) {
    __shared__ float s1[256], s2[256];
    int j = blockIdx.x;
    float a = 0.0f, b = 0.0f;
    for (int i = threadIdx.x; i < n; i += 256) {
        float v = x[(size_t)i * DH + j];
        a += v; b += v * v;
    }
    s1[threadIdx.x] = a; s2[threadIdx.x] = b;
    __syncthreads();
    for (int st = 128; st > 0; st >>= 1) {
        if (threadIdx.x < st) { s1[threadIdx.x] += s1[threadIdx.x + st];
                                s2[threadIdx.x] += s2[threadIdx.x + st]; }
        __syncthreads();
    }
    if (threadIdx.x == 0) {
        float m = s1[0] / (float)n;
        mu[j]  = m;
        var[j] = s2[0] / (float)n - m * m;
    }
}

__global__ void tgnn_bnapply(const float* __restrict__ x, const float* __restrict__ mu,
                             const float* __restrict__ var, const float* __restrict__ g,
                             const float* __restrict__ bt, float* __restrict__ y, int total) {
    int t = blockIdx.x * blockDim.x + threadIdx.x;
    if (t >= total) return;
    int j = t & (DH - 1);
    float v = g[j] * (x[t] - mu[j]) * rsqrtf(var[j] + 1e-5f) + bt[j];
    y[t] = fmaxf(v, 0.0f);
}

__global__ void tgnn_poolcnt(const int* __restrict__ batch, float* __restrict__ cnt, int n) {
    int t = blockIdx.x * blockDim.x + threadIdx.x;
    if (t < n) atomicAdd(&cnt[batch[t]], 1.0f);
}

__global__ void tgnn_poolsum(const float* __restrict__ h, const int* __restrict__ batch,
                             float* __restrict__ pooled, int total) {
    int t = blockIdx.x * blockDim.x + threadIdx.x;
    if (t >= total) return;
    int n = t >> 8, d = t & (DH - 1);
    atomicAdd(&pooled[batch[n] * DH + d], h[t]);
}

__global__ void tgnn_pooldiv(float* __restrict__ pooled, const float* __restrict__ cnt) {
    int t = blockIdx.x * blockDim.x + threadIdx.x;
    if (t >= GG * DH) return;
    pooled[t] /= fmaxf(cnt[t >> 8], 1.0f);
}

// out[g,o] = z1[g,:] @ W2[:,o] + b2[o]   (256 -> 10)
__global__ void tgnn_final(const float* __restrict__ z1, const float* __restrict__ W2,
                           const float* __restrict__ b2, float* __restrict__ out) {
    int t = blockIdx.x * blockDim.x + threadIdx.x;
    if (t >= GG * DOUT) return;
    int g = t / DOUT, o = t % DOUT;
    float acc = b2[o];
    #pragma unroll 8
    for (int c = 0; c < DH; ++c) acc += z1[g * DH + c] * W2[c * DOUT + o];
    out[t] = acc;
}

// ---------------------------------------------------------------------------
// Host orchestration
// ---------------------------------------------------------------------------
static inline void launch_gemm(const __bf16* A, const __bf16* Wt, const float* bias,
                               float* C, int M, int K, int Nout, int relu,
                               hipStream_t s) {
    dim3 grid(M / 32, Nout / 128);
    tgnn_gemm_wmma<<<grid, 256, 0, s>>>(A, Wt, bias, C, M, K, Nout, relu);
}

extern "C" void kernel_launch(void* const* d_in, const int* in_sizes, int n_in,
                              void* d_out, int out_size, void* d_ws, size_t ws_size,
                              hipStream_t stream) {
    const int N = in_sizes[0] / 128;       // 12800
    const int E = in_sizes[39] / 2;        // 102400

    const float* x  = (const float*)d_in[0];
    const float* ea = (const float*)d_in[1];
    const float* W1 = (const float*)d_in[35];
    const float* b1 = (const float*)d_in[36];
    const float* W2 = (const float*)d_in[37];
    const float* b2 = (const float*)d_in[38];
    const int*   ei = (const int*)d_in[39];
    const int*   batch = (const int*)d_in[40];
    const int* src = ei;
    const int* dst = ei + E;

    // workspace carve-up (256B aligned slabs)
    char* base = (char*)d_ws;
    size_t off = 0;
    auto alloc_f = [&](size_t n) {
        float* p = (float*)(base + off);
        off += ((n * 4 + 255) & ~(size_t)255);
        return p;
    };
    auto alloc_h = [&](size_t n) {
        __bf16* p = (__bf16*)(base + off);
        off += ((n * 2 + 255) & ~(size_t)255);
        return p;
    };

    float* qb   = alloc_f((size_t)N * DH);
    float* kb   = alloc_f((size_t)N * DH);
    float* vb   = alloc_f((size_t)N * DH);
    float* hA   = alloc_f((size_t)N * DH);    // post BN+ReLU layer output
    float* hB   = alloc_f((size_t)N * DH);    // conv accumulation buffer
    float* ep   = alloc_f((size_t)E * DH);    // edge projection
    float* alph = alloc_f((size_t)E * HH);
    float* amax = alloc_f((size_t)N * HH);
    float* ssum = alloc_f((size_t)N * HH);
    float* mu   = alloc_f(DH);
    float* var  = alloc_f(DH);
    float* pooled = alloc_f((size_t)GG * DH);
    float* cnt    = alloc_f(GG);
    float* z1     = alloc_f((size_t)GG * DH);
    __bf16* eabf = alloc_h((size_t)E * DEE);  // bf16 edge features (reused 3x)
    __bf16* xbf  = alloc_h((size_t)N * DH);   // bf16 layer input / pooled
    __bf16* wbf  = alloc_h((size_t)DEE * DH); // bf16 transposed weight scratch

    const int TB = 256;
    auto blocks = [&](int n) { return (n + TB - 1) / TB; };

    // edge features -> bf16 once (amortized over all 3 layers)
    tgnn_conv_bf16<<<blocks(E * DEE), TB, 0, stream>>>(ea, eabf, E * DEE);

    for (int l = 0; l < 3; ++l) {
        const int b0 = 2 + l * 11;
        const float* Wq = (const float*)d_in[b0 + 0];
        const float* bq = (const float*)d_in[b0 + 1];
        const float* Wk = (const float*)d_in[b0 + 2];
        const float* bk = (const float*)d_in[b0 + 3];
        const float* Wv = (const float*)d_in[b0 + 4];
        const float* bv = (const float*)d_in[b0 + 5];
        const float* We = (const float*)d_in[b0 + 6];
        const float* Ws = (const float*)d_in[b0 + 7];
        const float* bs = (const float*)d_in[b0 + 8];
        const float* gm = (const float*)d_in[b0 + 9];
        const float* bt = (const float*)d_in[b0 + 10];

        const float* xin = (l == 0) ? x : hA;
        const int Kin = (l == 0) ? 128 : DH;

        // node input -> bf16
        tgnn_conv_bf16<<<blocks(N * Kin), TB, 0, stream>>>(xin, xbf, N * Kin);

        // dense projections (WMMA); weights transposed+converted just-in-time
        tgnn_convtr<<<blocks(Kin * DH), TB, 0, stream>>>(Wq, wbf, Kin, DH);
        launch_gemm(xbf, wbf, bq, qb, N, Kin, DH, 0, stream);
        tgnn_convtr<<<blocks(Kin * DH), TB, 0, stream>>>(Wk, wbf, Kin, DH);
        launch_gemm(xbf, wbf, bk, kb, N, Kin, DH, 0, stream);
        tgnn_convtr<<<blocks(Kin * DH), TB, 0, stream>>>(Wv, wbf, Kin, DH);
        launch_gemm(xbf, wbf, bv, vb, N, Kin, DH, 0, stream);
        tgnn_convtr<<<blocks(DEE * DH), TB, 0, stream>>>(We, wbf, DEE, DH);
        launch_gemm(eabf, wbf, nullptr, ep, E, DEE, DH, 0, stream);
        // skip-connection term seeds the accumulation buffer
        tgnn_convtr<<<blocks(Kin * DH), TB, 0, stream>>>(Ws, wbf, Kin, DH);
        launch_gemm(xbf, wbf, bs, hB, N, Kin, DH, 0, stream);

        // attention scores + segment softmax
        tgnn_alpha<<<blocks(E * HH), TB, 0, stream>>>(qb, kb, ep, src, dst, alph, E);
        tgnn_fill<<<blocks(N * HH), TB, 0, stream>>>(amax, -1e30f, N * HH);
        tgnn_fill<<<blocks(N * HH), TB, 0, stream>>>(ssum, 0.0f, N * HH);
        tgnn_segmax<<<blocks(E * HH), TB, 0, stream>>>(alph, dst, amax, E);
        tgnn_expsum<<<blocks(E * HH), TB, 0, stream>>>(alph, dst, amax, ssum, E);

        // weighted message scatter into hB
        tgnn_scatter<<<E, TB, 0, stream>>>(vb, ep, alph, ssum, src, dst, hB);

        // batchnorm + relu -> hA
        tgnn_bnstats<<<DH, TB, 0, stream>>>(hB, mu, var, N);
        tgnn_bnapply<<<blocks(N * DH), TB, 0, stream>>>(hB, mu, var, gm, bt, hA, N * DH);
    }

    // mean pooling per graph
    tgnn_fill<<<blocks(GG * DH), TB, 0, stream>>>(pooled, 0.0f, GG * DH);
    tgnn_fill<<<blocks(GG), TB, 0, stream>>>(cnt, 0.0f, GG);
    tgnn_poolcnt<<<blocks(N), TB, 0, stream>>>(batch, cnt, N);
    tgnn_poolsum<<<blocks(N * DH), TB, 0, stream>>>(hA, batch, pooled, N * DH);
    tgnn_pooldiv<<<blocks(GG * DH), TB, 0, stream>>>(pooled, cnt);

    // MLP head: WMMA for 128x256x256, scalar kernel for 256->10
    tgnn_conv_bf16<<<blocks(GG * DH), TB, 0, stream>>>(pooled, xbf, GG * DH);
    tgnn_convtr<<<blocks(DH * DH), TB, 0, stream>>>(W1, wbf, DH, DH);
    launch_gemm(xbf, wbf, b1, z1, GG, DH, DH, 1, stream);
    tgnn_final<<<blocks(GG * DOUT), TB, 0, stream>>>(z1, W2, b2, (float*)d_out);
}
